// QuantizedLinear_8787503087749
// MI455X (gfx1250) — 8-bit matrix operand order repaired and checked by reading, compiled, not run on hardware
//
#include <hip/hip_runtime.h>

#define M_DIM 8192
#define K_DIM 4096
#define N_DIM 16384

#define BM 128
#define BN 128
#define BK 64
#define LDK 80
#define NBUF 3

typedef __attribute__((ext_vector_type(8))) int       v8i;
typedef __attribute__((ext_vector_type(4))) int       v4i;
typedef __attribute__((ext_vector_type(2))) int       v2i;
typedef __attribute__((ext_vector_type(8))) _Float16  v8h;

#define LDS_AS __attribute__((address_space(3)))
#define GLB_AS __attribute__((address_space(1)))

__device__ __forceinline__ void async_copy_b128(const void* g, void* l) {
#if __has_builtin(__builtin_amdgcn_global_load_async_to_lds_b128)
    __builtin_amdgcn_global_load_async_to_lds_b128(
        (GLB_AS v4i*)(uintptr_t)g, (LDS_AS v4i*)(uintptr_t)l, 0, 0);
#else
    *(v4i*)l = *(const v4i*)g;
#endif
}

#if __has_builtin(__builtin_amdgcn_s_wait_asynccnt)
#define WAIT_ASYNC(n) __builtin_amdgcn_s_wait_asynccnt(n)
#else
#define WAIT_ASYNC(n) asm volatile("s_wait_asynccnt %0" :: "i"(n) : "memory")
#endif

__global__ void quantize_kernel(const _Float16* __restrict__ x,
                                const float* __restrict__ act_scale,
                                const float* __restrict__ act_zp,
                                signed char* __restrict__ xq) {
    const float inv = 1.0f / act_scale[0];
    const float zp  = act_zp[0];
    const size_t base = ((size_t)blockIdx.x * blockDim.x + threadIdx.x) * 8;
    v8h h = *(const v8h*)(x + base);

    unsigned int packed[2] = {0u, 0u};
#pragma unroll
    for (int j = 0; j < 8; ++j) {
        float q = rintf((float)h[j] * inv) + zp;
        q = fminf(127.0f, fmaxf(-128.0f, q));
        int qi = (int)q;
        packed[j >> 2] |= ((unsigned int)(qi & 0xff)) << ((j & 3) * 8);
    }
    v2i st; st.x = (int)packed[0]; st.y = (int)packed[1];
    *(v2i*)(xq + base) = st;
}

__global__ void prep_kernel(const signed char* __restrict__ w,
                            const float* __restrict__ wscale,
                            const float* __restrict__ act_scale,
                            const float* __restrict__ act_zp,
                            const float* __restrict__ bias,
                            float* __restrict__ alpha,
                            float* __restrict__ beta) {
    const int n = blockIdx.x;
    const int t = threadIdx.x;
    const int* row = (const int*)(w + (size_t)n * K_DIM);

    int sum = 0;
#pragma unroll
    for (int i = 0; i < (K_DIM / 4) / 256; ++i) {
        int v = row[i * 256 + t];
        sum += (int)(signed char)(v)
             + (int)(signed char)(v >> 8)
             + (int)(signed char)(v >> 16)
             + (int)(signed char)(v >> 24);
    }

    __shared__ int red[256];
    red[t] = sum;
    __syncthreads();
#pragma unroll
    for (int s = 128; s > 0; s >>= 1) {
        if (t < s) red[t] += red[t + s];
        __syncthreads();
    }
    if (t == 0) {
        float a = wscale[n] * act_scale[0];
        alpha[n] = a;
        beta[n]  = bias[n] - (float)red[0] * act_zp[0] * a;
    }
}

__global__ void __launch_bounds__(256)
gemm_kernel(const signed char* __restrict__ Aq,
            const signed char* __restrict__ W,
            const float* __restrict__ alpha,
            const float* __restrict__ beta,
            _Float16* __restrict__ out) {
    __shared__ signed char As[NBUF][BM * LDK];
    __shared__ signed char Bs[NBUF][BN * LDK];

    const int tid    = threadIdx.x;
    const int lane   = tid & 31;
    const int wave   = tid >> 5;
    const int wm     = wave & 3;
    const int wn     = wave >> 2;
    const int ln     = lane & 15;
    const int laneHi = lane >> 4;

    const int bm = blockIdx.y * BM;
    const int bn = blockIdx.x * BN;

    auto issueTile = [&](int kt, int buf) {
#pragma unroll
        for (int c = 0; c < 2; ++c) {
            int idx = c * 256 + tid;
            int row = idx >> 2;
            int kc  = idx & 3;
            async_copy_b128(Aq + (size_t)(bm + row) * K_DIM + (size_t)kt * BK + kc * 16,
                            &As[buf][row * LDK + kc * 16]);
            async_copy_b128(W  + (size_t)(bn + row) * K_DIM + (size_t)kt * BK + kc * 16,
                            &Bs[buf][row * LDK + kc * 16]);
        }
    };

    v8i acc[2][4];
#pragma unroll
    for (int mi = 0; mi < 2; ++mi)
#pragma unroll
        for (int ni = 0; ni < 4; ++ni)
#pragma unroll
            for (int r = 0; r < 8; ++r) acc[mi][ni][r] = 0;

    float al[4], be[4];
#pragma unroll
    for (int ni = 0; ni < 4; ++ni) {
        const int col = bn + wn * 64 + ni * 16 + ln;
        al[ni] = alpha[col];
        be[ni] = beta[col];
    }

    const int NK = K_DIM / BK;
    issueTile(0, 0);
    issueTile(1, 1);

    for (int kt = 0; kt < NK; ++kt) {
        const int buf = kt % NBUF;

        if (kt + 1 < NK) WAIT_ASYNC(4);
        else             WAIT_ASYNC(0);
        __syncthreads();

        if (kt + 2 < NK) issueTile(kt + 2, (kt + 2) % NBUF);

        v8i afrag[2];
#pragma unroll
        for (int mi = 0; mi < 2; ++mi) {
            const int row = wm * 32 + mi * 16 + ln;
            const signed char* base = &As[buf][row * LDK + laneHi * 16];
            union { v8i v; v2i h[4]; } u;
            u.h[0] = *(const v2i*)(base + 0);
            u.h[1] = *(const v2i*)(base + 8);
            u.h[2] = *(const v2i*)(base + 32);
            u.h[3] = *(const v2i*)(base + 40);
            afrag[mi] = u.v;
        }

        v8i bfrag[4];
#pragma unroll
        for (int ni = 0; ni < 4; ++ni) {
            const int col = wn * 64 + ni * 16 + ln;
            const signed char* base = &Bs[buf][col * LDK + laneHi * 16];
            union { v8i v; v4i q[2]; } u;
            u.q[0] = *(const v4i*)(base + 0);
            u.q[1] = *(const v4i*)(base + 32);
            bfrag[ni] = u.v;
        }

#pragma unroll
        for (int mi = 0; mi < 2; ++mi)
#pragma unroll
            for (int ni = 0; ni < 4; ++ni)
                acc[mi][ni] = __builtin_amdgcn_wmma_i32_16x16x64_iu8(
                    true, afrag[mi], true, bfrag[ni], acc[mi][ni], false, false);
    }

#pragma unroll
    for (int ni = 0; ni < 4; ++ni) {
        const int col = bn + wn * 64 + ni * 16 + ln;
#pragma unroll
        for (int mi = 0; mi < 2; ++mi) {
            const int rbase = bm + wm * 32 + mi * 16 + laneHi * 8;
#pragma unroll
            for (int r = 0; r < 8; ++r) {
                float v = (float)acc[mi][ni][r] * al[ni] + be[ni];
                out[(size_t)(rbase + r) * N_DIM + col] = (_Float16)v;
            }
        }
    }
}

extern "C" void kernel_launch(void* const* d_in, const int* in_sizes, int n_in,
                              void* d_out, int out_size, void* d_ws, size_t ws_size,
                              hipStream_t stream) {
    const _Float16*    x      = (const _Float16*)d_in[0];
    const signed char* w      = (const signed char*)d_in[1];
    const float*       wscale = (const float*)d_in[2];
    const float*       ascale = (const float*)d_in[3];
    const float*       azp    = (const float*)d_in[4];
    const float*       bias   = (const float*)d_in[5];
    _Float16*          out    = (_Float16*)d_out;

    signed char* xq    = (signed char*)d_ws;
    float*       alpha = (float*)((char*)d_ws + (size_t)M_DIM * K_DIM);
    float*       beta  = alpha + N_DIM;

    quantize_kernel<<<(M_DIM * K_DIM) / (256 * 8), 256, 0, stream>>>(x, ascale, azp, xq);
    prep_kernel<<<N_DIM, 256, 0, stream>>>(w, wscale, ascale, azp, bias, alpha, beta);
    gemm_kernel<<<dim3(N_DIM / BN, M_DIM / BM), 256, 0, stream>>>(xq, w, alpha, beta, out);
}
